// Attention_3453153706123
// MI455X (gfx1250) — matrix operand order repaired and checked by reading, not compiled, not run on hardware
//
#include <hip/hip_runtime.h>
#include <hip/hip_bf16.h>
#include <stdint.h>

// MI455X (gfx1250) attention block.
// All GEMMs + attention matmuls use v_wmma_f32_16x16x32_bf16 (wave32).
// f32 inputs are converted to bf16 in-register while staging tiles into LDS.
// Attention K-tiles are staged with global_load_async_to_lds_b128 (ASYNCcnt
// pipeline, double-buffered).

typedef __attribute__((ext_vector_type(16))) __bf16 v16bf;
typedef __attribute__((ext_vector_type(8)))  __bf16 v8bf;
typedef __attribute__((ext_vector_type(4)))  __bf16 v4bf;
typedef __attribute__((ext_vector_type(8)))  float  v8f;

#define LDT 40   // padded LDS stride (halfs) for 32-wide bf16 tiles
#define LDK 72   // padded LDS stride (halfs) for 64-wide bf16 tiles

// ---------------------------------------------------------------------------
// GEMM: C[M,N] = A[M,K] * W[N,K]^T + bias   (W row-major N x K, torch-style)
// block tile 256(M) x 64(N), BK=32, 256 threads = 8 waves stacked along M,
// each wave owns a 32(M) x 64(N) patch = 2x4 WMMA accumulators
// (8 WMMAs per K-step vs 12 LDS fragment loads).
// ---------------------------------------------------------------------------
template<bool A_BF16, bool OUT_BF16>
__global__ __launch_bounds__(256)
void wmma_gemm(const void* __restrict__ Ap, const float* __restrict__ W,
               const float* __restrict__ bias, void* __restrict__ Cp,
               int M, int N, int K) {
  __shared__ __align__(16) __bf16 lA[256 * LDT];
  __shared__ __align__(16) __bf16 lB[64 * LDT];
  const int tid  = threadIdx.x;
  const int lane = tid & 31;
  const int w    = tid >> 5;      // wave id: rows w*32 .. w*32+31 of the block
  const int ln   = lane & 15;
  const int kh   = lane >> 4;     // half-wave selector
  const int bm   = blockIdx.y * 256;
  const int bn   = blockIdx.x * 64;

  v8f acc[2][4] = {};

  for (int kk = 0; kk < K; kk += 32) {
    __syncthreads();
    // ---- stage A tile (256 x 32) into LDS as bf16 ----
    if (A_BF16) {
      const __bf16* A = (const __bf16*)Ap;
      #pragma unroll
      for (int c = 0; c < 4; ++c) {
        int fid = tid + c * 256;                 // 1024 chunks of 8 halfs
        int r = fid >> 2, o = (fid & 3) * 8;
        v8bf d = *(const v8bf*)(A + (size_t)(bm + r) * K + kk + o);
        *(v8bf*)(lA + r * LDT + o) = d;
      }
    } else {
      const float* A = (const float*)Ap;
      #pragma unroll
      for (int c = 0; c < 8; ++c) {
        int fid = tid + c * 256;                 // 2048 chunks of 4 floats
        int r = fid >> 3, o = (fid & 7) * 4;
        float4 d = *(const float4*)(A + (size_t)(bm + r) * K + kk + o);
        v4bf hd = { (__bf16)d.x, (__bf16)d.y, (__bf16)d.z, (__bf16)d.w };
        *(v4bf*)(lA + r * LDT + o) = hd;
      }
    }
    // ---- stage B tile (64 rows of W x 32 cols) into LDS as bf16 ----
    #pragma unroll
    for (int c = 0; c < 2; ++c) {
      int fid = tid + c * 256;                   // 512 chunks of 4 floats
      int r = fid >> 3, o = (fid & 7) * 4;
      float4 d = *(const float4*)(W + (size_t)(bn + r) * K + kk + o);
      v4bf hd = { (__bf16)d.x, (__bf16)d.y, (__bf16)d.z, (__bf16)d.w };
      *(v4bf*)(lB + r * LDT + o) = hd;
    }
    // prefetch next K tile into L2 (global_prefetch_b8)
    if (kk + 32 < K) {
      __builtin_prefetch(W + (size_t)(bn + (tid >> 2)) * K + kk + 32, 0, 0);
      if (!A_BF16)
        __builtin_prefetch((const float*)Ap + (size_t)(bm + tid) * K + kk + 32, 0, 0);
    }
    __syncthreads();

    // ---- fragments (ISA 16-bit A / B VGPR layouts) ----
    v16bf af[2], bfb[4];
    #pragma unroll
    for (int mi = 0; mi < 2; ++mi) {
      const __bf16* p = lA + (w * 32 + mi * 16 + ln) * LDT;
      *(v8bf*)&af[mi]        = *(const v8bf*)(p + kh * 8);        // K 0..7 / 8..15
      *((v8bf*)&af[mi] + 1)  = *(const v8bf*)(p + 16 + kh * 8);   // K 16..23 / 24..31
    }
    #pragma unroll
    for (int ni = 0; ni < 4; ++ni) {
      const __bf16* p = lB + (ni * 16 + ln) * LDT;
      *(v8bf*)&bfb[ni]       = *(const v8bf*)(p + kh * 8);        // K 0..7 / 8..15
      *((v8bf*)&bfb[ni] + 1) = *(const v8bf*)(p + 16 + kh * 8);   // K 16..23 / 24..31
    }
    #pragma unroll
    for (int mi = 0; mi < 2; ++mi)
      #pragma unroll
      for (int ni = 0; ni < 4; ++ni)
        acc[mi][ni] = __builtin_amdgcn_wmma_f32_16x16x32_bf16(
            false, af[mi], false, bfb[ni], (short)0, acc[mi][ni], false, false);
  }

  // ---- epilogue: bias + store (C layout: VGPR v -> row v (+8 for hi half)) ----
  #pragma unroll
  for (int mi = 0; mi < 2; ++mi)
    #pragma unroll
    for (int ni = 0; ni < 4; ++ni) {
      int col  = bn + ni * 16 + ln;
      float b  = bias[col];
      int row0 = bm + w * 32 + mi * 16 + kh * 8;
      #pragma unroll
      for (int v = 0; v < 8; ++v) {
        float val = acc[mi][ni][v] + b;
        if (OUT_BF16)
          ((__bf16*)Cp)[(size_t)(row0 + v) * N + col] = (__bf16)val;
        else
          ((float*)Cp)[(size_t)(row0 + v) * N + col] = val;
      }
    }
}

// ---------------------------------------------------------------------------
// RoPE (rotate-half) applied in-place on bf16 q/k, layout (S, H*64).
// rope_cache row s: [cos(0..63) | sin(0..63)], stride 128.
// ---------------------------------------------------------------------------
__global__ void rope_kernel(__bf16* __restrict__ q, const float* __restrict__ cache,
                            int H, int total) {
  int i = blockIdx.x * blockDim.x + threadIdx.x;
  if (i >= total) return;
  int d = i & 31;
  int h = (i >> 5) % H;
  int s = i / (H * 32);
  const float* cr = cache + (size_t)s * 128;
  float c1 = cr[d], c2 = cr[d + 32];
  float s1 = cr[64 + d], s2 = cr[96 + d];
  __bf16* p = q + (size_t)s * H * 64 + h * 64 + d;
  float x1 = (float)p[0], x2 = (float)p[32];
  p[0]  = (__bf16)(x1 * c1 - x2 * s1);
  p[32] = (__bf16)(x2 * c2 + x1 * s2);
}

// ---------------------------------------------------------------------------
// Flash attention: grid (12 q-blocks, 64 heads), 256 threads = 8 waves.
// Wave w handles 16 query rows; per 32-key block: 4 WMMAs for Q.K^T,
// online softmax (half-wave shfl_xor reductions), P via LDS -> 4 WMMAs for P.V.
// K tiles staged with global_load_async_to_lds_b128, double-buffered on the
// ASYNCcnt counter (loads complete in order -> s_wait_asynccnt 1 releases the
// older buffer). GQA: kv head = h/8. Finalize: out *= sigmoid(lse-sink)/rowsum.
// ---------------------------------------------------------------------------
__global__ __launch_bounds__(256)
void attn_kernel(const __bf16* __restrict__ Q, const __bf16* __restrict__ Kb,
                 const __bf16* __restrict__ Vb, const float* __restrict__ sinks,
                 __bf16* __restrict__ O) {
  __shared__ __align__(16) __bf16 lK[2][32 * LDK];   // double-buffered K tiles
  __shared__ __align__(16) __bf16 lVt[64 * LDT];     // transposed V: d x keys
  __shared__ __align__(16) __bf16 lP[8 * 16 * LDT];  // per-wave P scratch
  const int tid  = threadIdx.x;
  const int lane = tid & 31;
  const int w    = tid >> 5;
  const int ln   = lane & 15;
  const int kh   = lane >> 4;
  const int qb   = blockIdx.x;
  const int h    = blockIdx.y;
  const int kvh  = h >> 3;               // GQA: n_rep = 8
  const int qw   = qb * 128 + w * 16;    // wave's first query row
  const int key  = tid >> 3;             // staging: this thread's key row
  const int o8   = (tid & 7) * 8;        //          and 8-half chunk

  // Q fragments: 16 rows x 64 d -> two 16x32 A-frags (kept in registers)
  v16bf qa[2];
  {
    const __bf16* qp = Q + (size_t)(qw + ln) * 4096 + h * 64;
    #pragma unroll
    for (int j = 0; j < 2; ++j) {
      *(v8bf*)&qa[j]       = *(const v8bf*)(qp + j * 32 + kh * 8);
      *((v8bf*)&qa[j] + 1) = *(const v8bf*)(qp + j * 32 + 16 + kh * 8);
    }
  }

  float rm[8], rs[8];
  v8f o[4] = {};
  #pragma unroll
  for (int v = 0; v < 8; ++v) { rm[v] = -1e30f; rs[v] = 0.f; }
  const float sink = sinks[h];
  const int nkb = qb * 4 + 4;            // causal: keys up to qb*128+127

  // async K stage: one b128 memory->LDS copy per lane, no VGPR round trip
  auto issueK = [&](int kb) {
    unsigned loff = (unsigned)(uintptr_t)(&lK[kb & 1][key * LDK + o8]);
    unsigned long long g = (unsigned long long)(uintptr_t)
        (Kb + ((size_t)(kb * 32 + key) * 8 + kvh) * 64 + o8);
    asm volatile("global_load_async_to_lds_b128 %0, %1, off"
                 :: "v"(loff), "v"(g) : "memory");
  };
  issueK(0);

  for (int kb = 0; kb < nkb; ++kb) {
    const int n0 = kb * 32;
    __syncthreads();                     // prev compute done -> buffers reusable
    if (kb + 1 < nkb) issueK(kb + 1);    // prefetch next K tile (async)
    {   // V: synchronous load + transpose into LDS
      v8bf vd = *(const v8bf*)(Vb + ((size_t)(n0 + key) * 8 + kvh) * 64 + o8);
      #pragma unroll
      for (int j = 0; j < 8; ++j) lVt[(o8 + j) * LDT + key] = vd[j];
    }
    if (kb + 1 < nkb)
      asm volatile("s_wait_asynccnt 0x1" ::: "memory");  // current K done, next in flight
    else
      asm volatile("s_wait_asynccnt 0x0" ::: "memory");
    __syncthreads();
    if (n0 > qw + 15) continue;          // block fully above this wave's rows

    const __bf16* lKc = lK[kb & 1];

    // ---- scores: S(16q x 32k) = Q . K^T, two 16x16 tiles, K-dim = 64 ----
    v8f s[2] = {};
    #pragma unroll
    for (int t = 0; t < 2; ++t)
      #pragma unroll
      for (int j = 0; j < 2; ++j) {
        v16bf kf;
        const __bf16* p = lKc + (t * 16 + ln) * LDK + j * 32;
        *(v8bf*)&kf       = *(const v8bf*)(p + kh * 8);
        *((v8bf*)&kf + 1) = *(const v8bf*)(p + 16 + kh * 8);
        s[t] = __builtin_amdgcn_wmma_f32_16x16x32_bf16(
            false, qa[j], false, kf, (short)0, s[t], false, false);
      }

    // ---- scale + causal mask + online softmax (rows = v + 8*kh) ----
    const int qrow_off = qw + 8 * kh;
    float pv0[8], pv1[8];
    #pragma unroll
    for (int v = 0; v < 8; ++v) {
      int qg = qrow_off + v;
      float s0 = s[0][v] * 0.125f;
      float s1 = s[1][v] * 0.125f;
      if (n0 + ln > qg)      s0 = -1e30f;
      if (n0 + 16 + ln > qg) s1 = -1e30f;
      float mx = fmaxf(s0, s1);
      #pragma unroll
      for (int m2 = 1; m2 <= 8; m2 <<= 1) mx = fmaxf(mx, __shfl_xor(mx, m2, 32));
      float nm = fmaxf(rm[v], mx);
      float cf = __expf(rm[v] - nm);
      float p0 = __expf(s0 - nm), p1 = __expf(s1 - nm);
      float sum = p0 + p1;
      #pragma unroll
      for (int m2 = 1; m2 <= 8; m2 <<= 1) sum += __shfl_xor(sum, m2, 32);
      rs[v] = rs[v] * cf + sum;
      rm[v] = nm;
      pv0[v] = p0; pv1[v] = p1;
      #pragma unroll
      for (int t = 0; t < 4; ++t) o[t][v] *= cf;
    }

    // ---- P: C-layout -> A-layout via per-wave LDS scratch ----
    __bf16* pw = lP + w * 16 * LDT;
    #pragma unroll
    for (int v = 0; v < 8; ++v) {
      pw[(v + 8 * kh) * LDT + ln]      = (__bf16)pv0[v];
      pw[(v + 8 * kh) * LDT + 16 + ln] = (__bf16)pv1[v];
    }
    v16bf pa;
    {
      const __bf16* pp = pw + ln * LDT;
      *(v8bf*)&pa       = *(const v8bf*)(pp + kh * 8);
      *((v8bf*)&pa + 1) = *(const v8bf*)(pp + 16 + kh * 8);
    }

    // ---- O(16 x 64) += P(16x32) . V(32x64): 4 WMMAs, K-dim = 32 keys ----
    #pragma unroll
    for (int t = 0; t < 4; ++t) {
      v16bf vf;
      const __bf16* p = lVt + (t * 16 + ln) * LDT;
      *(v8bf*)&vf       = *(const v8bf*)(p + kh * 8);
      *((v8bf*)&vf + 1) = *(const v8bf*)(p + 16 + kh * 8);
      o[t] = __builtin_amdgcn_wmma_f32_16x16x32_bf16(
          false, pa, false, vf, (short)0, o[t], false, false);
    }
  }

  // ---- finalize: softmax norm + attention-sink sigmoid rescale ----
  #pragma unroll
  for (int v = 0; v < 8; ++v) {
    float lse = rm[v] + __logf(rs[v]);
    float sg  = 1.f / (1.f + __expf(-(lse - sink)));
    float sc  = sg / rs[v];
    int row = qw + 8 * kh + v;
    #pragma unroll
    for (int t = 0; t < 4; ++t)
      O[(size_t)row * 4096 + h * 64 + t * 16 + ln] = (__bf16)(o[t][v] * sc);
  }
}

// ---------------------------------------------------------------------------
extern "C" void kernel_launch(void* const* d_in, const int* in_sizes, int n_in,
                              void* d_out, int out_size, void* d_ws, size_t ws_size,
                              hipStream_t stream) {
  const float* x     = (const float*)d_in[0];
  const float* rope  = (const float*)d_in[1];
  const float* wq_w  = (const float*)d_in[2];
  const float* wq_b  = (const float*)d_in[3];
  const float* wk_w  = (const float*)d_in[4];
  const float* wk_b  = (const float*)d_in[5];
  const float* wv_w  = (const float*)d_in[6];
  const float* wv_b  = (const float*)d_in[7];
  const float* wo_w  = (const float*)d_in[8];
  const float* wo_b  = (const float*)d_in[9];
  const float* sinks = (const float*)d_in[10];
  float* out = (float*)d_out;

  // workspace layout (bf16): q (S x 4096), k (S x 512), v (S x 512), attn (S x 4096)
  char* ws = (char*)d_ws;
  __bf16* qb = (__bf16*)ws;                                  // 12,582,912 B
  __bf16* kb = (__bf16*)(ws + 12582912);                     //  1,572,864 B
  __bf16* vb = (__bf16*)(ws + 12582912 + 1572864);           //  1,572,864 B
  __bf16* ab = (__bf16*)(ws + 12582912 + 2 * 1572864);       // 12,582,912 B

  dim3 blk(256);
  // QKV projections (f32 in, bf16 out): M=1536 (6 row-blocks of 256), K=2880
  wmma_gemm<false, true><<<dim3(64, 6), blk, 0, stream>>>(x, wq_w, wq_b, qb, 1536, 4096, 2880);
  wmma_gemm<false, true><<<dim3(8,  6), blk, 0, stream>>>(x, wk_w, wk_b, kb, 1536, 512,  2880);
  wmma_gemm<false, true><<<dim3(8,  6), blk, 0, stream>>>(x, wv_w, wv_b, vb, 1536, 512,  2880);
  // RoPE on q (H=64) and k (Hkv=8)
  rope_kernel<<<12288, 256, 0, stream>>>(qb, rope, 64, 1536 * 64 * 32);
  rope_kernel<<<1536,  256, 0, stream>>>(kb, rope, 8,  1536 * 8 * 32);
  // causal GQA flash attention with sink rescale
  attn_kernel<<<dim3(12, 64), blk, 0, stream>>>(qb, kb, vb, sinks, ab);
  // output projection (bf16 in, f32 out): K=4096, N=2880
  wmma_gemm<true, false><<<dim3(45, 6), blk, 0, stream>>>(ab, wo_w, wo_b, out, 1536, 2880, 4096);
}
